// AdaptiveFilterAttention_48009144434987
// MI455X (gfx1250) — matrix operand order repaired and checked by reading, not compiled, not run on hardware
//
#include <hip/hip_runtime.h>
#include <hip/hip_fp16.h>

typedef __attribute__((ext_vector_type(16))) _Float16 v16h;
typedef __attribute__((ext_vector_type(8)))  float    v8f;
typedef __attribute__((ext_vector_type(4)))  int      v4i;

#define D_DIM 1024
#define T_SEQ 2048
#define B_SZ  2
#define NH    16
#define HD    64
#define NTOK  (B_SZ * T_SEQ)   // 4096

// LDS row strides (halves), padded +8 to avoid bank conflicts; keep 16B align.
#define KT_STRIDE 40   // 32-key rows  (80 B)
#define VT_STRIDE 72   // 64-hd rows   (144 B)
#define WT_STRIDE 72   // 64-n rows    (144 B)
#define PL_STRIDE 40   // 32-key rows  (80 B)

union V16 {
  v16h  h;
  uint4 u[2];
};

// ---------------------------------------------------------------------------
// Async global->LDS 16-byte copy (ASYNCcnt path), with sync fallback.
// Probed signature: (as1 v4i* src, as3 v4i* dst, imm offset, imm cpol)
// ---------------------------------------------------------------------------
#if __has_builtin(__builtin_amdgcn_global_load_async_to_lds_b128)
#define AFA_ASYNC 1
#else
#define AFA_ASYNC 0
#endif

__device__ __forceinline__ void afa_cp16(const _Float16* g, _Float16* l) {
#if AFA_ASYNC
  __builtin_amdgcn_global_load_async_to_lds_b128(
      (__attribute__((address_space(1))) v4i*)g,
      (__attribute__((address_space(3))) v4i*)l, 0, 0);
#else
  *(uint4*)l = *(const uint4*)g;
#endif
}

__device__ __forceinline__ void afa_wait_async() {
#if AFA_ASYNC
#if __has_builtin(__builtin_amdgcn_s_wait_asynccnt)
  __builtin_amdgcn_s_wait_asynccnt(0);
#else
  asm volatile("s_wait_asynccnt 0x0" ::: "memory");
#endif
#endif
}

// ---------------------------------------------------------------------------
// Prep: f32 -> f16 conversion of x
// ---------------------------------------------------------------------------
__global__ void afa_convert_x(const float* __restrict__ x,
                              _Float16* __restrict__ xh, int n) {
  int i = blockIdx.x * blockDim.x + threadIdx.x;
  if (i < n) xh[i] = (_Float16)x[i];
}

// Prep: Wt[k*D + n] = (f16) W[n*D + k]  (LDS-tiled transpose)
__global__ void afa_transpose_w(const float* __restrict__ W,
                                _Float16* __restrict__ Wt) {
  __shared__ float tile[32][33];
  int x = blockIdx.x * 32 + threadIdx.x;
  int y = blockIdx.y * 32 + threadIdx.y;
  #pragma unroll
  for (int j = 0; j < 32; j += 8)
    tile[threadIdx.y + j][threadIdx.x] = W[(size_t)(y + j) * D_DIM + x];
  __syncthreads();
  int xt = blockIdx.y * 32 + threadIdx.x;
  int yt = blockIdx.x * 32 + threadIdx.y;
  #pragma unroll
  for (int j = 0; j < 32; j += 8)
    Wt[(size_t)(yt + j) * D_DIM + xt] = (_Float16)tile[threadIdx.x][threadIdx.y + j];
}

// ---------------------------------------------------------------------------
// Projection GEMM: 4 waves/block, M=64 (16 rows/wave), N=64.
// Weight tile (32k x 64n) staged to LDS with async copies, double buffered.
// kstore=1 writes K transposed [b][h][hd][T].
// ---------------------------------------------------------------------------
__global__ __launch_bounds__(128) void afa_proj_gemm(
    const _Float16* __restrict__ A, const _Float16* __restrict__ Wt,
    const float* __restrict__ bias, _Float16* __restrict__ outh,
    float* __restrict__ outf, int kstore) {
  __shared__ alignas(16) _Float16 wtile[2][32 * WT_STRIDE];
  const int tid  = threadIdx.x;
  const int wave = tid >> 5;
  const int lane = tid & 31;
  const int half = lane >> 4;
  const int lrow = lane & 15;
  const int m0 = blockIdx.x * 64 + wave * 16;
  const int n0 = blockIdx.y * 64;

  v8f acc[4];
  #pragma unroll
  for (int nt = 0; nt < 4; ++nt)
    #pragma unroll
    for (int i = 0; i < 8; ++i) acc[nt][i] = 0.0f;

  const _Float16* aRow = A + (size_t)(m0 + lrow) * D_DIM;

  auto stageW = [&](int buf, int kbase) {
    #pragma unroll
    for (int i = 0; i < 2; ++i) {      // 256 16B chunks over 128 threads
      int c  = tid + i * 128;
      int k  = c >> 3;
      int n8 = (c & 7) * 8;
      afa_cp16(Wt + (size_t)(kbase + k) * D_DIM + n0 + n8,
               &wtile[buf][k * WT_STRIDE + n8]);
    }
  };

  stageW(0, 0);
  afa_wait_async();
  __syncthreads();

  for (int kb = 0; kb < D_DIM; kb += 32) {
    const int cur = (kb >> 5) & 1;
    if (kb + 32 < D_DIM) stageW(cur ^ 1, kb + 32);

    V16 a;
    const _Float16* ap = aRow + kb + half * 8;
    a.u[0] = *(const uint4*)(ap);
    a.u[1] = *(const uint4*)(ap + 16);
    __builtin_prefetch(aRow + kb + 64, 0, 1);
    #pragma unroll
    for (int nt = 0; nt < 4; ++nt) {
      V16 b;
      const _Float16* bp = &wtile[cur][nt * 16 + lrow];
      #pragma unroll
      for (int j = 0; j < 16; ++j)
        b.h[j] = bp[((j & 7) + 8 * half + 16 * (j >> 3)) * WT_STRIDE];
      acc[nt] = __builtin_amdgcn_wmma_f32_16x16x32_f16(
          false, a.h, false, b.h, (short)0, acc[nt], false, false);
    }
    afa_wait_async();
    __syncthreads();
  }

  #pragma unroll
  for (int nt = 0; nt < 4; ++nt) {
    int col = n0 + nt * 16 + lrow;
    float bv = bias[col];
    #pragma unroll
    for (int r = 0; r < 8; ++r) {
      int row = m0 + r + 8 * half;
      float v = acc[nt][r] + bv;
      if (outf) {
        outf[(size_t)row * D_DIM + col] = v;
      } else if (kstore) {
        int b  = row >> 11;            // / T_SEQ
        int t  = row & (T_SEQ - 1);
        int hh = col >> 6;
        int kd = col & 63;
        outh[(size_t)((b * NH + hh) * HD + kd) * T_SEQ + t] = (_Float16)v;
      } else {
        outh[(size_t)row * D_DIM + col] = (_Float16)v;
      }
    }
  }
}

// ---------------------------------------------------------------------------
// Flash attention with exp(-alpha*|i-j|) decay folded into scores.
// 4 waves/block = 64 queries share 32-key K/V tiles staged in LDS (async).
// ---------------------------------------------------------------------------
__global__ __launch_bounds__(128) void afa_attention(
    const _Float16* __restrict__ Qh,  // [NTOK][D]
    const _Float16* __restrict__ Kt,  // [B][H][HD][T]
    const _Float16* __restrict__ Vh,  // [NTOK][D]
    _Float16* __restrict__ Ah,        // [NTOK][D]
    const float* __restrict__ alphap) {
  __shared__ alignas(16) _Float16 ktile[64 * KT_STRIDE];      // [hd][key]
  __shared__ alignas(16) _Float16 vtile[32 * VT_STRIDE];      // [key][hd]
  __shared__ alignas(16) _Float16 plds[4][16 * PL_STRIDE];    // per-wave P
  const int tid  = threadIdx.x;
  const int wave = tid >> 5;
  const int lane = tid & 31;
  const int half = lane >> 4;
  const int lrow = lane & 15;
  const int q0 = blockIdx.x * 64 + wave * 16;
  const int h  = blockIdx.y;
  const int b  = blockIdx.z;
  const float alpha = alphap[0];
  const float scale = 0.125f;  // hd^-0.5

  // Q as two 16x32 A-tiles over the 64-wide head dim (per-wave rows)
  const _Float16* qrow = Qh + (size_t)(b * T_SEQ + q0 + lrow) * D_DIM + h * HD;
  V16 qa[2];
  #pragma unroll
  for (int kk = 0; kk < 2; ++kk) {
    const _Float16* qp = qrow + kk * 32 + half * 8;
    qa[kk].u[0] = *(const uint4*)(qp);
    qa[kk].u[1] = *(const uint4*)(qp + 16);
  }

  const _Float16* ktBase = Kt + (size_t)((b * NH + h) * HD) * T_SEQ;
  const _Float16* vBase  = Vh + (size_t)(b * T_SEQ) * D_DIM + h * HD;

  v8f oacc[4];
  float m[8], l[8];
  #pragma unroll
  for (int r = 0; r < 8; ++r) { m[r] = -1e30f; l[r] = 0.0f; }
  #pragma unroll
  for (int nt = 0; nt < 4; ++nt)
    #pragma unroll
    for (int i = 0; i < 8; ++i) oacc[nt][i] = 0.0f;

  for (int kb = 0; kb < T_SEQ; kb += 32) {
    __syncthreads();  // previous-iteration readers done before restage

    // ---- stage K tile [64 hd][32 keys] and V tile [32 keys][64 hd] ----
    #pragma unroll
    for (int i = 0; i < 2; ++i) {      // 256 chunks of 16B over 128 threads
      int c  = tid + i * 128;
      int kd = c >> 2;
      int c8 = (c & 3) * 8;
      afa_cp16(ktBase + (size_t)kd * T_SEQ + kb + c8,
               &ktile[kd * KT_STRIDE + c8]);
    }
    #pragma unroll
    for (int i = 0; i < 2; ++i) {
      int c   = tid + i * 128;
      int key = c >> 3;
      int c8  = (c & 7) * 8;
      afa_cp16(vBase + (size_t)(kb + key) * D_DIM + c8,
               &vtile[key * VT_STRIDE + c8]);
    }
    afa_wait_async();
    __syncthreads();

    // ---- S = Q K^T for 32 keys (two 16x16 C tiles) ----
    v8f sacc[2];
    #pragma unroll
    for (int s = 0; s < 2; ++s)
      #pragma unroll
      for (int i = 0; i < 8; ++i) sacc[s][i] = 0.0f;

    #pragma unroll
    for (int kk = 0; kk < 2; ++kk) {
      const _Float16* kp = &ktile[(kk * 32) * KT_STRIDE + lrow];
      #pragma unroll
      for (int s = 0; s < 2; ++s) {
        V16 bmat;
        #pragma unroll
        for (int j = 0; j < 16; ++j)
          bmat.h[j] = kp[((j & 7) + 8 * half + 16 * (j >> 3)) * KT_STRIDE + s * 16];
        sacc[s] = __builtin_amdgcn_wmma_f32_16x16x32_f16(
            false, qa[kk].h, false, bmat.h, (short)0, sacc[s], false, false);
      }
    }

    // ---- scale * exp(-alpha*|i-j|) on raw scores ----
    #pragma unroll
    for (int s = 0; s < 2; ++s)
      #pragma unroll
      for (int r = 0; r < 8; ++r) {
        float qi = (float)(q0 + r + 8 * half);
        float kj = (float)(kb + s * 16 + lrow);
        sacc[s][r] *= scale * __expf(-alpha * fabsf(qi - kj));
      }

    // ---- online softmax (row reductions across each 16-lane half) ----
    #pragma unroll
    for (int r = 0; r < 8; ++r) {
      float v = fmaxf(sacc[0][r], sacc[1][r]);
      #pragma unroll
      for (int off = 1; off < 16; off <<= 1) v = fmaxf(v, __shfl_xor(v, off, 32));
      float mnew = fmaxf(m[r], v);
      float sc = __expf(m[r] - mnew);
      m[r] = mnew;
      l[r] *= sc;
      #pragma unroll
      for (int nt = 0; nt < 4; ++nt) oacc[nt][r] *= sc;
    }
    #pragma unroll
    for (int s = 0; s < 2; ++s)
      #pragma unroll
      for (int r = 0; r < 8; ++r)
        sacc[s][r] = __expf(sacc[s][r] - m[r]);
    #pragma unroll
    for (int r = 0; r < 8; ++r) {
      float v = sacc[0][r] + sacc[1][r];
      #pragma unroll
      for (int off = 1; off < 16; off <<= 1) v += __shfl_xor(v, off, 32);
      l[r] += v;
    }

    // ---- P: C-layout -> A-layout via per-wave LDS region ----
    #pragma unroll
    for (int s = 0; s < 2; ++s)
      #pragma unroll
      for (int r = 0; r < 8; ++r)
        plds[wave][(r + 8 * half) * PL_STRIDE + s * 16 + lrow] =
            (_Float16)sacc[s][r];
    V16 pa;
    {
      const _Float16* pp = &plds[wave][lrow * PL_STRIDE + half * 8];
      pa.u[0] = *(const uint4*)(pp);
      pa.u[1] = *(const uint4*)(pp + 16);
    }

    // ---- O += P @ V (V from LDS) ----
    const _Float16* vp = &vtile[lrow];
    #pragma unroll
    for (int nt = 0; nt < 4; ++nt) {
      V16 bmat;
      #pragma unroll
      for (int j = 0; j < 16; ++j)
        bmat.h[j] = vp[((j & 7) + 8 * half + 16 * (j >> 3)) * VT_STRIDE + nt * 16];
      oacc[nt] = __builtin_amdgcn_wmma_f32_16x16x32_f16(
          false, pa.h, false, bmat.h, (short)0, oacc[nt], false, false);
    }
  }

  // ---- normalize and store f16 attention output ----
  #pragma unroll
  for (int nt = 0; nt < 4; ++nt)
    #pragma unroll
    for (int r = 0; r < 8; ++r) {
      int row = b * T_SEQ + q0 + r + 8 * half;
      int col = h * HD + nt * 16 + lrow;
      Ah[(size_t)row * D_DIM + col] = (_Float16)(oacc[nt][r] / l[r]);
    }
}

// ---------------------------------------------------------------------------
extern "C" void kernel_launch(void* const* d_in, const int* in_sizes, int n_in,
                              void* d_out, int out_size, void* d_ws, size_t ws_size,
                              hipStream_t stream) {
  const float* x     = (const float*)d_in[0];
  const float* Wq    = (const float*)d_in[1];
  const float* bq    = (const float*)d_in[2];
  const float* Wk    = (const float*)d_in[3];
  const float* bk    = (const float*)d_in[4];
  const float* Wv    = (const float*)d_in[5];
  const float* bv    = (const float*)d_in[6];
  const float* Wo    = (const float*)d_in[7];
  const float* bo    = (const float*)d_in[8];
  const float* alpha = (const float*)d_in[9];
  float* out = (float*)d_out;
  (void)in_sizes; (void)n_in; (void)out_size; (void)ws_size;

  char* ws = (char*)d_ws;
  size_t off = 0;
  const size_t act_bytes = (size_t)NTOK * D_DIM * sizeof(_Float16);   // 8 MB
  const size_t w_bytes   = (size_t)D_DIM * D_DIM * sizeof(_Float16);  // 2 MB
  _Float16* xh  = (_Float16*)(ws + off); off += act_bytes;
  _Float16* Qh  = (_Float16*)(ws + off); off += act_bytes;
  _Float16* Kt  = (_Float16*)(ws + off); off += act_bytes;
  _Float16* Vh  = (_Float16*)(ws + off); off += act_bytes;
  _Float16* Ah  = (_Float16*)(ws + off); off += act_bytes;
  _Float16* Wqt = (_Float16*)(ws + off); off += w_bytes;
  _Float16* Wkt = (_Float16*)(ws + off); off += w_bytes;
  _Float16* Wvt = (_Float16*)(ws + off); off += w_bytes;
  _Float16* Wot = (_Float16*)(ws + off); off += w_bytes;

  // Prep: x -> f16, weights -> transposed f16
  {
    int n = NTOK * D_DIM;
    afa_convert_x<<<(n + 255) / 256, 256, 0, stream>>>(x, xh, n);
    dim3 tb(32, 8), tg(D_DIM / 32, D_DIM / 32);
    afa_transpose_w<<<tg, tb, 0, stream>>>(Wq, Wqt);
    afa_transpose_w<<<tg, tb, 0, stream>>>(Wk, Wkt);
    afa_transpose_w<<<tg, tb, 0, stream>>>(Wv, Wvt);
    afa_transpose_w<<<tg, tb, 0, stream>>>(Wo, Wot);
  }

  // QKV projections: 64x64 tiles per block (4 waves)
  {
    dim3 g(NTOK / 64, D_DIM / 64);
    afa_proj_gemm<<<g, 128, 0, stream>>>(xh, Wqt, bq, Qh, nullptr, 0);
    afa_proj_gemm<<<g, 128, 0, stream>>>(xh, Wkt, bk, Kt, nullptr, 1);
    afa_proj_gemm<<<g, 128, 0, stream>>>(xh, Wvt, bv, Vh, nullptr, 0);
  }

  // Attention with exponential-decay weighting: 64 queries per block
  {
    dim3 g(T_SEQ / 64, NH, B_SZ);
    afa_attention<<<g, 128, 0, stream>>>(Qh, Kt, Vh, Ah, alpha);
  }

  // Output projection (f32 epilogue straight to d_out)
  {
    dim3 g(NTOK / 64, D_DIM / 64);
    afa_proj_gemm<<<g, 128, 0, stream>>>(Ah, Wot, bo, nullptr, out, 0);
  }
}
